// Decoder_21758304322268
// MI455X (gfx1250) — matrix operand order repaired and checked by reading, not compiled, not run on hardware
//
#include <hip/hip_runtime.h>

// ---------------------------------------------------------------------------
// MI455X (gfx1250) decoder inference.
//  - All large matmuls use v_wmma_f32_16x16x32_bf16. Weights converted once
//    per launch to bf16 (~65MB, resident in 192MB L2 across the 32 steps).
//  - Activations kept in bf16 by their producer kernels -> GEMM inner loop is
//    pure global_load_b128 + 4x v_wmma (16x64 output per wave, A reused).
//  - f32 accumulate everywhere; LN/softmax/loss in f32.
//  - Deterministic: fixed-order LDS tree reductions, no float atomics.
// ---------------------------------------------------------------------------

typedef __attribute__((ext_vector_type(16))) __bf16  v16bf;
typedef __attribute__((ext_vector_type(8)))  float   v8f;
typedef __attribute__((ext_vector_type(4)))  unsigned int u32x4;

#define HID   1024
#define EMB   512
#define VOC   50000
#define CUT0  3000
#define CUT1  20000
#define BATCH 128
#define SEQ   128
#define TLEN  32
#define BN_INV 0.9999950000374997f

__device__ __forceinline__ float sigmoidf_(float x) { return 1.0f / (1.0f + __expf(-x)); }
__device__ __forceinline__ unsigned short to_bf16_(float x) {
  return __builtin_bit_cast(unsigned short, (__bf16)x);
}
__device__ __forceinline__ v16bf load_bf16x16_(const unsigned short* p) {
  u32x4 lo = *(const u32x4*)(p);        // 8 bf16
  u32x4 hi = *(const u32x4*)(p + 16);   // 8 bf16 at +16 elements
  typedef __attribute__((ext_vector_type(8))) __bf16 v8bf;
  v8bf a = __builtin_bit_cast(v8bf, lo);
  v8bf b = __builtin_bit_cast(v8bf, hi);
  v16bf r;
#pragma unroll
  for (int j = 0; j < 8; ++j) { r[j] = a[j]; r[8 + j] = b[j]; }
  return r;
}

// ------------------------- weight f32 -> bf16 (padded) ----------------------
__global__ __launch_bounds__(256) void cvt_bf16_pad(const float* __restrict__ src,
                                                    unsigned short* __restrict__ dst,
                                                    int N, int K, long long total) {
  long long i = (long long)blockIdx.x * 256 + threadIdx.x;
  if (i >= total) return;
  long long n = i / K, k = i - n * K;
  dst[i] = (n < N) ? to_bf16_(src[n * (long long)K + k]) : (unsigned short)0;
}

// ------------------------- activation f32 -> bf16 ---------------------------
__global__ __launch_bounds__(256) void cvt_act(const float* __restrict__ src,
                                               unsigned short* __restrict__ dst, int n) {
  int i = blockIdx.x * 256 + threadIdx.x;
  if (i < n) dst[i] = to_bf16_(src[i]);
}

// ------------------------- WMMA GEMM: C = A * W^T + bias --------------------
// A: (128, K) bf16 row-major.  W: (Npad, K) bf16 row-major, Npad % 64 == 0.
// C: (128, N) f32 row-major (ld = N).  One wave -> 16x64 tile (4 WMMA/K-step).
__global__ __launch_bounds__(256) void gemm_bf16(const unsigned short* __restrict__ A,
                                                 const unsigned short* __restrict__ W,
                                                 const float* __restrict__ bias,
                                                 float* __restrict__ C,
                                                 int N, int Npad, int K) {
  const int lane = threadIdx.x & 31;
  const int wave = threadIdx.x >> 5;
  const int nT0  = (blockIdx.x * 8 + wave) * 4;     // first of 4 n-tiles
  if (nT0 * 16 >= Npad) return;
  const int mBase = blockIdx.y * 16;
  const int half  = lane >> 4;     // 0: lanes 0-15, 1: lanes 16-31
  const int l15   = lane & 15;

  // A (16-bit 16x32 layout): row=l15, K = {kk..kk+7, kk+16..kk+23}, kk = kb + 8*half
  const unsigned short* pa = A + (size_t)(mBase + l15) * K + (half << 3);
  // B (32x16): col=l15, K = {kk..kk+7, kk+16..kk+23}, kk = kb + 8*half (same as A)
  const unsigned short* pb0 = W + (size_t)((nT0 + 0) * 16 + l15) * K + (half << 3);
  const unsigned short* pb1 = W + (size_t)((nT0 + 1) * 16 + l15) * K + (half << 3);
  const unsigned short* pb2 = W + (size_t)((nT0 + 2) * 16 + l15) * K + (half << 3);
  const unsigned short* pb3 = W + (size_t)((nT0 + 3) * 16 + l15) * K + (half << 3);

  v8f acc0 = {}, acc1 = {}, acc2 = {}, acc3 = {};
  for (int kb = 0; kb < K; kb += 32) {
    __builtin_prefetch(pb0 + kb + 512, 0, 1);       // global_prefetch_b8
    v16bf av = load_bf16x16_(pa + kb);
    v16bf b0 = load_bf16x16_(pb0 + kb);
    v16bf b1 = load_bf16x16_(pb1 + kb);
    v16bf b2 = load_bf16x16_(pb2 + kb);
    v16bf b3 = load_bf16x16_(pb3 + kb);
    acc0 = __builtin_amdgcn_wmma_f32_16x16x32_bf16(false, av, false, b0, (short)0, acc0, false, false);
    acc1 = __builtin_amdgcn_wmma_f32_16x16x32_bf16(false, av, false, b1, (short)0, acc1, false, false);
    acc2 = __builtin_amdgcn_wmma_f32_16x16x32_bf16(false, av, false, b2, (short)0, acc2, false, false);
    acc3 = __builtin_amdgcn_wmma_f32_16x16x32_bf16(false, av, false, b3, (short)0, acc3, false, false);
  }
#pragma unroll
  for (int j = 0; j < 4; ++j) {
    v8f acc = (j == 0) ? acc0 : (j == 1) ? acc1 : (j == 2) ? acc2 : acc3;
    int col = (nT0 + j) * 16 + l15;
    if (col < N) {
      float bval = bias ? bias[col] : 0.0f;
#pragma unroll
      for (int r = 0; r < 8; ++r) {
        int m = mBase + r + (half << 3);
        C[(size_t)m * N + col] = acc[r] + bval;
      }
    }
  }
}

// ------------------------- LayerNorm per 1024-chunk (3 chunks) --------------
// grid = (3, rows); in-place safe (src may equal dst).
__global__ __launch_bounds__(256) void ln3_kernel(const float* __restrict__ src,
                                                  float* __restrict__ dst,
                                                  const float* __restrict__ g,
                                                  const float* __restrict__ bc) {
  const int chunk = blockIdx.x, b = blockIdx.y, tid = threadIdx.x;
  const float* x = src + (size_t)b * 3072 + chunk * 1024;
  float*       y = dst + (size_t)b * 3072 + chunk * 1024;
  float v[4], s = 0.f, s2 = 0.f;
#pragma unroll
  for (int j = 0; j < 4; ++j) { v[j] = x[tid + j * 256]; s += v[j]; s2 += v[j] * v[j]; }
  __shared__ float r1[256], r2[256];
  r1[tid] = s; r2[tid] = s2; __syncthreads();
  for (int o = 128; o > 0; o >>= 1) {
    if (tid < o) { r1[tid] += r1[tid + o]; r2[tid] += r2[tid + o]; }
    __syncthreads();
  }
  float mu  = r1[0] * (1.0f / 1024.0f);
  float var = r2[0] * (1.0f / 1024.0f) - mu * mu;
  float inv = rsqrtf(var + 1e-5f);
#pragma unroll
  for (int j = 0; j < 4; ++j) {
    int i = tid + j * 256;
    y[i] = (v[j] - mu) * inv * g[chunk * 1024 + i] + bc[chunk * 1024 + i];
  }
}

// ------------------------- GRU gate combine (writes f32 h + bf16 h) ---------
__global__ __launch_bounds__(256) void gru_combine(const float* __restrict__ gi, int giStride,
                                                   const float* __restrict__ gh,
                                                   float* __restrict__ h,
                                                   unsigned short* __restrict__ hbf) {
  int idx = blockIdx.x * 256 + threadIdx.x;
  if (idx >= BATCH * HID) return;
  int b = idx >> 10, j = idx & 1023;
  const float* gib = gi + (size_t)b * giStride;
  const float* ghb = gh + (size_t)b * 3072;
  float r = sigmoidf_(gib[j] + ghb[j]);
  float z = sigmoidf_(gib[1024 + j] + ghb[1024 + j]);
  float n = tanhf(gib[2048 + j] + r * ghb[2048 + j]);
  float hn = (1.0f - z) * n + z * h[idx];
  h[idx] = hn;
  hbf[idx] = to_bf16_(hn);
}

// ------------------------- attention: pt scalar -----------------------------
__global__ __launch_bounds__(256) void pt_kernel(const float* __restrict__ ptmp,
                                                 const float* __restrict__ w2,
                                                 const float* __restrict__ b2,
                                                 float* __restrict__ pt) {
  int b = blockIdx.x, tid = threadIdx.x;
  float v = tanhf(ptmp[b * 256 + tid]) * w2[tid];
  __shared__ float red[256];
  red[tid] = v; __syncthreads();
  for (int o = 128; o > 0; o >>= 1) { if (tid < o) red[tid] += red[tid + o]; __syncthreads(); }
  if (tid == 0) pt[b] = (float)SEQ * sigmoidf_(red[0] + b2[0]);
}

// ------------------------- attention: scores --------------------------------
__global__ __launch_bounds__(64) void score_kernel(const float* __restrict__ ctx,
                                                   const float* __restrict__ wa,
                                                   const int* __restrict__ inputs,
                                                   float* __restrict__ score) {
  int s = blockIdx.x, b = blockIdx.y, tid = threadIdx.x;
  const float* hv = ctx + ((size_t)s * BATCH + b) * 2048;
  const float* wv = wa + (size_t)b * 2048;
  float acc = 0.f;
  for (int d = tid; d < 2048; d += 64) acc += hv[d] * wv[d];
  __shared__ float red[64];
  red[tid] = acc; __syncthreads();
  for (int o = 32; o > 0; o >>= 1) { if (tid < o) red[tid] += red[tid + o]; __syncthreads(); }
  if (tid == 0) {
    float m = (inputs[b * SEQ + s] == 0) ? -1e9f : 0.0f;
    score[b * SEQ + s] = red[0] + m;
  }
}

// ------------------------- attention: masked softmax * gaussian -------------
__global__ __launch_bounds__(128) void attn_softmax(float* __restrict__ score,
                                                    const float* __restrict__ pt) {
  int b = blockIdx.x, s = threadIdx.x;
  float v = score[b * SEQ + s];
  __shared__ float red[128];
  red[s] = v; __syncthreads();
  for (int o = 64; o > 0; o >>= 1) { if (s < o) red[s] = fmaxf(red[s], red[s + o]); __syncthreads(); }
  float m = red[0]; __syncthreads();
  float e = __expf(v - m);
  red[s] = e; __syncthreads();
  for (int o = 64; o > 0; o >>= 1) { if (s < o) red[s] += red[s + o]; __syncthreads(); }
  float d = (float)s - pt[b];
  score[b * SEQ + s] = (e / red[0]) * __expf(-d * d / 18.0f);
}

// ------------------------- attention: context (bf16 out for GEMM) -----------
__global__ __launch_bounds__(256) void attn_context(const float* __restrict__ ctx,
                                                    const float* __restrict__ At,
                                                    unsigned short* __restrict__ out) {
  int b = blockIdx.y, d = blockIdx.x * 256 + threadIdx.x;
  __shared__ float at[SEQ];
  if (threadIdx.x < SEQ) at[threadIdx.x] = At[b * SEQ + threadIdx.x];
  __syncthreads();
  float acc = 0.f;
  for (int s = 0; s < SEQ; ++s) acc += at[s] * ctx[((size_t)s * BATCH + b) * 2048 + d];
  out[(size_t)b * 2048 + d] = to_bf16_(acc);
}

// ------------------------- output nonlinearity (bf16 out) -------------------
__global__ __launch_bounds__(256) void y_kernel(const float* __restrict__ fwout,
                                                const float* __restrict__ bn_g,
                                                const float* __restrict__ bn_b,
                                                unsigned short* __restrict__ ybf) {
  int idx = blockIdx.x * 256 + threadIdx.x;
  if (idx >= BATCH * HID) return;
  int j = idx & 1023;
  ybf[idx] = to_bf16_(tanhf(fwout[idx] * BN_INV * bn_g[j] + bn_b[j]));
}

// ------------------------- log-sum-exp per row ------------------------------
__global__ __launch_bounds__(256) void lse_kernel(const float* __restrict__ z, int N,
                                                  float* __restrict__ lse) {
  int b = blockIdx.x, tid = threadIdx.x;
  const float* row = z + (size_t)b * N;
  float m = -3.4e38f;
  for (int i = tid; i < N; i += 256) m = fmaxf(m, row[i]);
  __shared__ float red[256];
  red[tid] = m; __syncthreads();
  for (int o = 128; o > 0; o >>= 1) { if (tid < o) red[tid] = fmaxf(red[tid], red[tid + o]); __syncthreads(); }
  float M = red[0]; __syncthreads();
  float s = 0.f;
  for (int i = tid; i < N; i += 256) s += __expf(row[i] - M);
  red[tid] = s; __syncthreads();
  for (int o = 128; o > 0; o >>= 1) { if (tid < o) red[tid] += red[tid + o]; __syncthreads(); }
  if (tid == 0) lse[b] = M + __logf(red[0]);
}

// ------------------------- argmax over full vocab + target logprob ----------
__global__ __launch_bounds__(256) void pick_kernel(const float* __restrict__ zh,
                                                   const float* __restrict__ z1,
                                                   const float* __restrict__ z2,
                                                   const float* __restrict__ lseh,
                                                   const float* __restrict__ lse1,
                                                   const float* __restrict__ lse2,
                                                   const int* __restrict__ targets, int t,
                                                   float* __restrict__ preds_out,
                                                   float* __restrict__ loss_out) {
  int b = blockIdx.x, tid = threadIdx.x;
  float lh = lseh[b], l1 = lse1[b], l2 = lse2[b];
  const float* rh = zh + (size_t)b * 3002;
  const float* r1 = z1 + (size_t)b * 17000;
  const float* r2 = z2 + (size_t)b * 30000;
  float adj1 = rh[CUT0] - lh;
  float adj2 = rh[CUT0 + 1] - lh;
  float best = -3.4e38f; int bi = 0x7fffffff;
  for (int i = tid; i < CUT0; i += 256) {
    float v = rh[i] - lh;
    if (v > best) { best = v; bi = i; }
  }
  for (int i = tid; i < CUT1 - CUT0; i += 256) {
    float v = r1[i] - l1 + adj1; int gidx = CUT0 + i;
    if (v > best || (v == best && gidx < bi)) { best = v; bi = gidx; }
  }
  for (int i = tid; i < VOC - CUT1; i += 256) {
    float v = r2[i] - l2 + adj2; int gidx = CUT1 + i;
    if (v > best || (v == best && gidx < bi)) { best = v; bi = gidx; }
  }
  __shared__ float bv[256]; __shared__ int bix[256];
  bv[tid] = best; bix[tid] = bi; __syncthreads();
  for (int o = 128; o > 0; o >>= 1) {
    if (tid < o) {
      float ov = bv[tid + o]; int oi = bix[tid + o];
      if (ov > bv[tid] || (ov == bv[tid] && oi < bix[tid])) { bv[tid] = ov; bix[tid] = oi; }
    }
    __syncthreads();
  }
  if (tid == 0) {
    preds_out[b] = (float)bix[0];
    int tgt = targets[b * TLEN + t];
    float lp;
    if (tgt < CUT0)      lp = rh[tgt] - lh;
    else if (tgt < CUT1) lp = r1[tgt - CUT0] - l1 + adj1;
    else                 lp = r2[tgt - CUT1] - l2 + adj2;
    loss_out[b] = lp;
  }
}

// ------------------------- embedding gather (bf16 out) ----------------------
__global__ __launch_bounds__(256) void emb_gather(const float* __restrict__ emb,
                                                  const float* __restrict__ predsF,
                                                  unsigned short* __restrict__ prev_y) {
  int b = blockIdx.x;
  int p = (int)predsF[b];
  for (int e = threadIdx.x; e < EMB; e += 256)
    prev_y[(size_t)b * EMB + e] = to_bf16_(emb[(size_t)p * EMB + e]);
}

// ------------------------- final deterministic loss reduce ------------------
__global__ __launch_bounds__(256) void loss_final(const float* __restrict__ lbuf,
                                                  float* __restrict__ out) {
  int tid = threadIdx.x;
  float s = 0.f;
  for (int i = tid; i < TLEN * BATCH; i += 256) s += lbuf[i];
  __shared__ float red[256];
  red[tid] = s; __syncthreads();
  for (int o = 128; o > 0; o >>= 1) { if (tid < o) red[tid] += red[tid + o]; __syncthreads(); }
  if (tid == 0) out[0] = -red[0] / ((float)BATCH * (float)TLEN);
}

// ===========================================================================
// Host side
// ===========================================================================
#define IN_CTX 0
#define IN_HIDDEN 1
#define IN_INPUTS 2
#define IN_TARGETS 3
#define CELL(c, f) (5 + (c) * 8 + (f))   // f: 0 Wi,1 bi,2 Wh,3 bh,4 lni_g,5 lni_b,6 lnh_g,7 lnh_b
#define IN_EMB 37
#define IN_PTW1 38
#define IN_PTB1 39
#define IN_PTW2 40
#define IN_PTB2 41
#define IN_WAW 42
#define IN_WAB 43
#define IN_FWW 44
#define IN_FWB 45
#define IN_BNG 46
#define IN_BNB 47
#define IN_HEADW 48
#define IN_HEADB 49
#define IN_P1 50
#define IN_T1W 51
#define IN_T1B 52
#define IN_P2 53
#define IN_T2W 54
#define IN_T2B 55

static inline void launch_cvt(const float* src, unsigned short* dst, int N, int Npad, int K,
                              hipStream_t s) {
  long long total = (long long)Npad * K;
  int blocks = (int)((total + 255) / 256);
  cvt_bf16_pad<<<blocks, 256, 0, s>>>(src, dst, N, K, total);
}

static inline void launch_gemm(const unsigned short* A, const unsigned short* W,
                               const float* bias, float* C, int N, int Npad, int K,
                               hipStream_t s) {
  dim3 grid((Npad / 64 + 7) / 8, 8);   // 8 waves/block, 4 n-tiles/wave
  gemm_bf16<<<grid, 256, 0, s>>>(A, W, bias, C, N, Npad, K);
}

extern "C" void kernel_launch(void* const* d_in, const int* in_sizes, int n_in,
                              void* d_out, int out_size, void* d_ws, size_t ws_size,
                              hipStream_t stream) {
  (void)in_sizes; (void)n_in; (void)out_size; (void)ws_size;
  char* ws = (char*)d_ws;
  size_t off = 0;
  auto alloc = [&](size_t bytes) -> void* {
    off = (off + 255) & ~(size_t)255;
    void* p = ws + off;
    off += bytes;
    return p;
  };

  // ---- bf16 weight buffers (rows padded to multiples of 64) ----
  unsigned short* wWi0  = (unsigned short*)alloc((size_t)3072 * 512 * 2);
  unsigned short* wWh[4];
  for (int c = 0; c < 4; ++c) wWh[c] = (unsigned short*)alloc((size_t)3072 * 1024 * 2);
  unsigned short* wWi1  = (unsigned short*)alloc((size_t)3072 * 2048 * 2);
  unsigned short* wPt1  = (unsigned short*)alloc((size_t)256 * 1024 * 2);
  unsigned short* wWa   = (unsigned short*)alloc((size_t)2048 * 1024 * 2);
  unsigned short* wFw   = (unsigned short*)alloc((size_t)1024 * 1024 * 2);
  unsigned short* wHead = (unsigned short*)alloc((size_t)3072 * 1024 * 2);
  unsigned short* wP1   = (unsigned short*)alloc((size_t)256 * 1024 * 2);
  unsigned short* wT1   = (unsigned short*)alloc((size_t)17024 * 256 * 2);
  unsigned short* wP2   = (unsigned short*)alloc((size_t)64 * 1024 * 2);
  unsigned short* wT2   = (unsigned short*)alloc((size_t)30016 * 64 * 2);

  // ---- bf16 activation buffers (GEMM A operands) ----
  unsigned short* hbf    = (unsigned short*)alloc((size_t)BATCH * HID * 2);
  unsigned short* prevbf = (unsigned short*)alloc((size_t)BATCH * EMB * 2);
  unsigned short* cvbf   = (unsigned short*)alloc((size_t)BATCH * 2048 * 2);
  unsigned short* ybf    = (unsigned short*)alloc((size_t)BATCH * HID * 2);
  unsigned short* p1bf   = (unsigned short*)alloc((size_t)BATCH * 256 * 2);
  unsigned short* p2bf   = (unsigned short*)alloc((size_t)BATCH * 64 * 2);

  // ---- f32 activation buffers ----
  float* h      = (float*)alloc((size_t)BATCH * HID * 4);
  float* gi     = (float*)alloc((size_t)BATCH * 3072 * 4);
  float* gh     = (float*)alloc((size_t)BATCH * 3072 * 4);
  float* giC2   = (float*)alloc(3072 * 4);
  float* giC3   = (float*)alloc(3072 * 4);
  float* ptmp   = (float*)alloc((size_t)BATCH * 256 * 4);
  float* ptv    = (float*)alloc(BATCH * 4);
  float* wa     = (float*)alloc((size_t)BATCH * 2048 * 4);
  float* score  = (float*)alloc((size_t)BATCH * SEQ * 4);
  float* fwout  = (float*)alloc((size_t)BATCH * HID * 4);
  float* p1f    = (float*)alloc((size_t)BATCH * 256 * 4);
  float* p2f    = (float*)alloc((size_t)BATCH * 64 * 4);
  float* zh     = (float*)alloc((size_t)BATCH * 3008 * 4);
  float* z1     = (float*)alloc((size_t)BATCH * 17024 * 4);
  float* z2     = (float*)alloc((size_t)BATCH * 30016 * 4);
  float* lseh   = (float*)alloc(BATCH * 4);
  float* lse1   = (float*)alloc(BATCH * 4);
  float* lse2   = (float*)alloc(BATCH * 4);
  float* lbuf   = (float*)alloc((size_t)TLEN * BATCH * 4);

  const float* ctx     = (const float*)d_in[IN_CTX];
  const int*   inputs  = (const int*)d_in[IN_INPUTS];
  const int*   targets = (const int*)d_in[IN_TARGETS];
  float*       outF    = (float*)d_out;

  // ---- one-time per launch: weight conversion (reused for 32 steps) ----
  launch_cvt((const float*)d_in[CELL(0, 0)], wWi0, 3072, 3072, 512, stream);
  for (int c = 0; c < 4; ++c)
    launch_cvt((const float*)d_in[CELL(c, 2)], wWh[c], 3072, 3072, 1024, stream);
  launch_cvt((const float*)d_in[CELL(1, 0)], wWi1, 3072, 3072, 2048, stream);
  launch_cvt((const float*)d_in[IN_PTW1], wPt1, 256, 256, 1024, stream);
  launch_cvt((const float*)d_in[IN_WAW], wWa, 2048, 2048, 1024, stream);
  launch_cvt((const float*)d_in[IN_FWW], wFw, 1024, 1024, 1024, stream);
  launch_cvt((const float*)d_in[IN_HEADW], wHead, 3002, 3072, 1024, stream);
  launch_cvt((const float*)d_in[IN_P1], wP1, 256, 256, 1024, stream);
  launch_cvt((const float*)d_in[IN_T1W], wT1, 17000, 17024, 256, stream);
  launch_cvt((const float*)d_in[IN_P2], wP2, 64, 64, 1024, stream);
  launch_cvt((const float*)d_in[IN_T2W], wT2, 30000, 30016, 64, stream);

  // cells 2/3: zero input path (Wi on zero vector) -> gi = ln3(bi), constant
  ln3_kernel<<<dim3(3, 1), 256, 0, stream>>>((const float*)d_in[CELL(2, 1)], giC2,
                                             (const float*)d_in[CELL(2, 4)],
                                             (const float*)d_in[CELL(2, 5)]);
  ln3_kernel<<<dim3(3, 1), 256, 0, stream>>>((const float*)d_in[CELL(3, 1)], giC3,
                                             (const float*)d_in[CELL(3, 4)],
                                             (const float*)d_in[CELL(3, 5)]);

  // initial state
  hipMemcpyAsync(h, d_in[IN_HIDDEN], (size_t)BATCH * HID * 4, hipMemcpyDeviceToDevice, stream);
  cvt_act<<<(BATCH * HID + 255) / 256, 256, 0, stream>>>((const float*)d_in[IN_HIDDEN], hbf,
                                                         BATCH * HID);
  hipMemsetAsync(prevbf, 0, (size_t)BATCH * EMB * 2, stream);   // bf16 zero == 0x0000

  const int EW = (BATCH * HID + 255) / 256;  // elementwise grid over (B,H)

  for (int t = 0; t < TLEN; ++t) {
    // ---- GRU cell 0: input prev_y (B,512) ----
    launch_gemm(prevbf, wWi0, (const float*)d_in[CELL(0, 1)], gi, 3072, 3072, 512, stream);
    ln3_kernel<<<dim3(3, BATCH), 256, 0, stream>>>(gi, gi, (const float*)d_in[CELL(0, 4)],
                                                   (const float*)d_in[CELL(0, 5)]);
    launch_gemm(hbf, wWh[0], (const float*)d_in[CELL(0, 3)], gh, 3072, 3072, 1024, stream);
    ln3_kernel<<<dim3(3, BATCH), 256, 0, stream>>>(gh, gh, (const float*)d_in[CELL(0, 6)],
                                                   (const float*)d_in[CELL(0, 7)]);
    gru_combine<<<EW, 256, 0, stream>>>(gi, 3072, gh, h, hbf);

    // ---- attention on ht = h ----
    launch_gemm(hbf, wPt1, (const float*)d_in[IN_PTB1], ptmp, 256, 256, 1024, stream);
    pt_kernel<<<BATCH, 256, 0, stream>>>(ptmp, (const float*)d_in[IN_PTW2],
                                         (const float*)d_in[IN_PTB2], ptv);
    launch_gemm(hbf, wWa, (const float*)d_in[IN_WAB], wa, 2048, 2048, 1024, stream);
    score_kernel<<<dim3(SEQ, BATCH), 64, 0, stream>>>(ctx, wa, inputs, score);
    attn_softmax<<<BATCH, 128, 0, stream>>>(score, ptv);
    attn_context<<<dim3(8, BATCH), 256, 0, stream>>>(ctx, score, cvbf);

    // ---- GRU cell 1: input context (B,2048) ----
    launch_gemm(cvbf, wWi1, (const float*)d_in[CELL(1, 1)], gi, 3072, 3072, 2048, stream);
    ln3_kernel<<<dim3(3, BATCH), 256, 0, stream>>>(gi, gi, (const float*)d_in[CELL(1, 4)],
                                                   (const float*)d_in[CELL(1, 5)]);
    launch_gemm(hbf, wWh[1], (const float*)d_in[CELL(1, 3)], gh, 3072, 3072, 1024, stream);
    ln3_kernel<<<dim3(3, BATCH), 256, 0, stream>>>(gh, gh, (const float*)d_in[CELL(1, 6)],
                                                   (const float*)d_in[CELL(1, 7)]);
    gru_combine<<<EW, 256, 0, stream>>>(gi, 3072, gh, h, hbf);

    // ---- GRU cells 2,3: constant input path ----
    for (int c = 2; c < 4; ++c) {
      launch_gemm(hbf, wWh[c], (const float*)d_in[CELL(c, 3)], gh, 3072, 3072, 1024, stream);
      ln3_kernel<<<dim3(3, BATCH), 256, 0, stream>>>(gh, gh, (const float*)d_in[CELL(c, 6)],
                                                     (const float*)d_in[CELL(c, 7)]);
      gru_combine<<<EW, 256, 0, stream>>>((c == 2) ? giC2 : giC3, 0, gh, h, hbf);
    }

    // ---- output head ----
    launch_gemm(hbf, wFw, (const float*)d_in[IN_FWB], fwout, 1024, 1024, 1024, stream);
    y_kernel<<<EW, 256, 0, stream>>>(fwout, (const float*)d_in[IN_BNG],
                                     (const float*)d_in[IN_BNB], ybf);

    launch_gemm(ybf, wHead, (const float*)d_in[IN_HEADB], zh, 3002, 3072, 1024, stream);
    launch_gemm(ybf, wP1, nullptr, p1f, 256, 256, 1024, stream);
    cvt_act<<<(BATCH * 256 + 255) / 256, 256, 0, stream>>>(p1f, p1bf, BATCH * 256);
    launch_gemm(p1bf, wT1, (const float*)d_in[IN_T1B], z1, 17000, 17024, 256, stream);
    launch_gemm(ybf, wP2, nullptr, p2f, 64, 64, 1024, stream);
    cvt_act<<<(BATCH * 64 + 255) / 256, 256, 0, stream>>>(p2f, p2bf, BATCH * 64);
    launch_gemm(p2bf, wT2, (const float*)d_in[IN_T2B], z2, 30000, 30016, 64, stream);

    lse_kernel<<<BATCH, 256, 0, stream>>>(zh, 3002, lseh);
    lse_kernel<<<BATCH, 256, 0, stream>>>(z1, 17000, lse1);
    lse_kernel<<<BATCH, 256, 0, stream>>>(z2, 30000, lse2);

    pick_kernel<<<BATCH, 256, 0, stream>>>(zh, z1, z2, lseh, lse1, lse2, targets, t,
                                           outF + (size_t)t * BATCH, lbuf + (size_t)t * BATCH);
    emb_gather<<<BATCH, 256, 0, stream>>>((const float*)d_in[IN_EMB],
                                          outF + (size_t)t * BATCH, prevbf);
  }

  loss_final<<<1, 256, 0, stream>>>(lbuf, outF + (size_t)TLEN * BATCH);
}
